// GATLayer_32461362823479
// MI455X (gfx1250) — matrix operand order repaired and checked by reading, not compiled, not run on hardware
//
#include <hip/hip_runtime.h>

typedef __attribute__((ext_vector_type(16))) __bf16 v16bf;
typedef __attribute__((ext_vector_type(8)))  float  v8f;
typedef unsigned int u32;
typedef u32 u32x4 __attribute__((ext_vector_type(4)));
typedef u32 u32x8 __attribute__((ext_vector_type(8)));

static constexpr int   N_NODES = 8192;
static constexpr int   F_IN    = 256;
static constexpr int   F_OUT   = 128;
static constexpr float NEG_INF = -9.0e15f;

static constexpr int CHUNK = 128;  // adj columns moved per TDM descriptor
static constexpr int ROWS  = 32;   // adj rows per wave (two 16-row strips)

// ---------------------------------------------------------------------------
// Kernel 1: WhT[n][m] = (h @ W.T)[m][n] in bf16.  One wave per 16x16 tile.
// A operand layout (16-bit A 16x32): lane m=L%16, half=L>>4;
//   elems 0..7  -> K = half*8 + e ;  elems 8..15 -> K = 16 + half*8 + e
// B operand layout (16-bit B 32x16): lane n=L%16, khalf=L>>4;
//   elems 0..7  -> K = khalf*8 + e ;  elems 8..15 -> K = 16 + khalf*8 + e  (same as A)
// ---------------------------------------------------------------------------
__global__ __launch_bounds__(32) void wh_kernel(const float* __restrict__ h,
                                                const float* __restrict__ W,
                                                __bf16* __restrict__ whT) {
  const int tile  = blockIdx.x;
  const int strip = tile >> 3;
  const int nt    = tile & 7;
  const int m0 = strip * 16, n0 = nt * 16;
  const int lane = threadIdx.x & 31;
  const int m = lane & 15, half = lane >> 4;

  v8f c = {};
  const float* hrow = h + (size_t)(m0 + m) * F_IN;
  const float* wrow = W + (size_t)(n0 + m) * F_IN;

  for (int k0 = 0; k0 < F_IN; k0 += 32) {
    v16bf A, B;
#pragma unroll
    for (int e = 0; e < 8; ++e) {
      A[e]     = (__bf16)hrow[k0 + half * 8 + e];
      A[e + 8] = (__bf16)hrow[k0 + 16 + half * 8 + e];
    }
#pragma unroll
    for (int e = 0; e < 16; ++e)
      B[e] = (__bf16)wrow[k0 + (e < 8 ? half * 8 + e : 16 + half * 8 + (e - 8))];
    c = __builtin_amdgcn_wmma_f32_16x16x32_bf16(false, A, false, B,
                                                (short)0, c, false, false);
  }
#pragma unroll
  for (int v = 0; v < 8; ++v)
    whT[(size_t)(n0 + m) * N_NODES + m0 + v + 8 * half] = (__bf16)c[v];
}

// ---------------------------------------------------------------------------
// Kernel 1b: s_src / s_dst GEMVs, coalesced over WhT columns.
// ---------------------------------------------------------------------------
__global__ void s_kernel(const __bf16* __restrict__ whT,
                         const float* __restrict__ a,
                         float* __restrict__ s_src, float* __restrict__ s_dst) {
  const int i = blockIdx.x * blockDim.x + threadIdx.x;
  if (i >= N_NODES) return;
  float ss = 0.f, sd = 0.f;
#pragma unroll 4
  for (int n = 0; n < F_OUT; ++n) {
    const float w = (float)whT[(size_t)n * N_NODES + i];
    ss += w * a[n];
    sd += w * a[F_OUT + n];
  }
  s_src[i] = ss;
  s_dst[i] = sd;
}

// ---------------------------------------------------------------------------
// Kernel 2: flash-style fused masked-softmax(P) @ Wh.
// One wave handles two 16-row strips (32 rows) so each WhT B-tile load feeds
// two WMMAs.  adj tiles (32 x 128 int32) are DMA'd global->LDS by the Tensor
// Data Mover (double-buffered, s_wait_tensorcnt synchronized).
// ---------------------------------------------------------------------------
__global__ __launch_bounds__(32) void gat_flash_kernel(
    const int* __restrict__ adj, const __bf16* __restrict__ whT,
    const float* __restrict__ s_src, const float* __restrict__ s_dst,
    float* __restrict__ accbuf, float2* __restrict__ stats, int S) {
  __shared__ int smem[2 * ROWS * CHUNK];  // 2 x 16KB double buffer

  const int bid  = blockIdx.x;
  const int seg  = bid % S;
  const int pair = bid / S;  // 0..255
  const int i0   = pair * ROWS;
  const int lane = threadIdx.x & 31;
  const int m = lane & 15, half = lane >> 4;

  const int jlen    = N_NODES / S;  // multiple of CHUNK for S in {1,2,4,8}
  const int j_begin = seg * jlen;
  const int nchunks = jlen / CHUNK;

  // --- Tensor DMA descriptor, group 1 (invariant across chunks) ---
  u32x8 g1;
  g1[0] = 0x00020000u;                      // data_size = 4B; no iterate/pad
  g1[1] = 0xFFFF0000u;                      // tensor_dim0 = 0x7FFFFFFF (lo16<<16)
  g1[2] = 0xFFFF7FFFu;                      // dim0 hi16 | tensor_dim1 lo16
  g1[3] = ((u32)CHUNK << 16) | 0x7FFFu;     // dim1 hi16 | tile_dim0 = CHUNK
  g1[4] = (u32)ROWS;                        // tile_dim1 = ROWS, tile_dim2 = 0
  g1[5] = (u32)N_NODES;                     // tensor_dim0_stride (elements)
  g1[6] = 0u;
  g1[7] = 0u;

  const uint64_t gaddr0 =
      (uint64_t)(uintptr_t)(adj + (size_t)i0 * N_NODES + j_begin);
  const u32 lds_base = (u32)(uintptr_t)(void*)smem;

  auto issue_tdm = [&](int c) {
    const uint64_t ga = gaddr0 + (uint64_t)c * (CHUNK * 4);
    u32x4 g0;
    g0[0] = 1u;  // count=1 valid user descriptor, no gather
    g0[1] = lds_base + (u32)((c & 1) * (ROWS * CHUNK * 4));
    g0[2] = (u32)ga;
    g0[3] = ((u32)(ga >> 32) & 0x01FFFFFFu) | 0x80000000u;  // addr hi | type=2
    asm volatile("tensor_load_to_lds %0, %1" ::"s"(g0), "s"(g1) : "memory");
  };

  const float ssrc[2] = {s_src[i0 + m], s_src[i0 + 16 + m]};
  float run_max[2] = {NEG_INF, NEG_INF};
  float run_sum[2] = {0.0f, 0.0f};
  v8f acc[2][8] = {};

  // A-operand K offsets for this lane (pairs start here)
  int koff[8];
#pragma unroll
  for (int r = 0; r < 8; ++r)
    koff[r] = 2 * r + half * 16;

  issue_tdm(0);
  for (int c = 0; c < nchunks; ++c) {
    __builtin_amdgcn_s_wait_tensorcnt(0);      // chunk c resident in LDS
    if (c + 1 < nchunks) issue_tdm(c + 1);     // overlap next DMA with compute
    const int* lbuf = smem + (c & 1) * (ROWS * CHUNK);

    for (int jl = 0; jl < CHUNK; jl += 32) {
      const int j0 = j_begin + c * CHUNK + jl;

      // s_dst tile (shared by both strips), in A-element order
      float d[16];
#pragma unroll
      for (int r = 0; r < 8; ++r) {
        const int k = koff[r];
        d[2 * r]     = s_dst[j0 + k];
        d[2 * r + 1] = s_dst[j0 + k + 1];
      }

      v16bf A[2];
#pragma unroll
      for (int p = 0; p < 2; ++p) {
        const int* lrow = lbuf + (p * 16 + m) * CHUNK + jl;
        float e[16];
        float emax = NEG_INF;
#pragma unroll
        for (int r = 0; r < 8; ++r) {
          const int k  = koff[r];
          const int a0 = lrow[k], a1 = lrow[k + 1];  // ds_load_b64
          float x0 = ssrc[p] + d[2 * r];
          float x1 = ssrc[p] + d[2 * r + 1];
          x0 = (x0 > 0.0f) ? x0 : 0.2f * x0;
          x1 = (x1 > 0.0f) ? x1 : 0.2f * x1;
          const float e0 = (a0 > 0) ? x0 : NEG_INF;
          const float e1 = (a1 > 0) ? x1 : NEG_INF;
          e[2 * r] = e0; e[2 * r + 1] = e1;
          emax = fmaxf(emax, fmaxf(e0, e1));
        }
        emax = fmaxf(emax, __shfl_xor(emax, 16, 32));
        const float new_max = fmaxf(run_max[p], emax);
        const float alpha   = __expf(run_max[p] - new_max);

        float psum = 0.0f;
#pragma unroll
        for (int k = 0; k < 16; ++k) {
          const float pv = __expf(e[k] - new_max);  // finite NEG_INF => ref-exact
          psum += pv;
          A[p][k] = (__bf16)pv;
        }
        psum += __shfl_xor(psum, 16, 32);
        run_sum[p] = run_sum[p] * alpha + psum;
        run_max[p] = new_max;

        // rescale accumulators: C-layout row M = v + 8*half, alpha in lane M
        float asc[8];
#pragma unroll
        for (int v = 0; v < 8; ++v) asc[v] = __shfl(alpha, v + 8 * half, 32);
#pragma unroll
        for (int t = 0; t < 8; ++t)
#pragma unroll
          for (int v = 0; v < 8; ++v) acc[p][t][v] *= asc[v];
      }

      // shared B tiles: each WhT load feeds both strips' WMMAs
#pragma unroll
      for (int t = 0; t < 8; ++t) {
        const __bf16* bptr =
            whT + (size_t)(t * 16 + m) * N_NODES + j0 + half * 16;
        const v16bf B = *(const v16bf*)bptr;  // 32B contiguous, L2-resident
        acc[0][t] = __builtin_amdgcn_wmma_f32_16x16x32_bf16(
            false, A[0], false, B, (short)0, acc[0][t], false, false);
        acc[1][t] = __builtin_amdgcn_wmma_f32_16x16x32_bf16(
            false, A[1], false, B, (short)0, acc[1][t], false, false);
      }
    }
  }

  // store unnormalized partials + (max,sum) stats
#pragma unroll
  for (int p = 0; p < 2; ++p) {
#pragma unroll
    for (int t = 0; t < 8; ++t)
#pragma unroll
      for (int v = 0; v < 8; ++v) {
        const int row = i0 + p * 16 + v + 8 * half;
        accbuf[((size_t)seg * N_NODES + row) * F_OUT + t * 16 + m] =
            acc[p][t][v];
      }
    if (half == 0)
      stats[(size_t)seg * N_NODES + i0 + p * 16 + m] =
          make_float2(run_max[p], run_sum[p]);
  }
}

// ---------------------------------------------------------------------------
// Kernel 3: merge S partial flash segments and normalize.
// ---------------------------------------------------------------------------
__global__ void combine_kernel(const float* __restrict__ accbuf,
                               const float2* __restrict__ stats,
                               float* __restrict__ out, int S) {
  const int idx = blockIdx.x * blockDim.x + threadIdx.x;
  if (idx >= N_NODES * F_OUT) return;
  const int i = idx / F_OUT;
  float M = NEG_INF;
  for (int s = 0; s < S; ++s) M = fmaxf(M, stats[(size_t)s * N_NODES + i].x);
  float denom = 0.f, val = 0.f;
  for (int s = 0; s < S; ++s) {
    const float2 st = stats[(size_t)s * N_NODES + i];
    const float  w  = __expf(st.x - M);
    denom += st.y * w;
    val   += accbuf[(size_t)s * N_NODES * F_OUT + idx] * w;
  }
  out[idx] = val / denom;
}

// ---------------------------------------------------------------------------
extern "C" void kernel_launch(void* const* d_in, const int* in_sizes, int n_in,
                              void* d_out, int out_size, void* d_ws,
                              size_t ws_size, hipStream_t stream) {
  const float* h   = (const float*)d_in[0];
  const int*   adj = (const int*)d_in[1];
  const float* W   = (const float*)d_in[2];
  const float* a   = (const float*)d_in[3];
  float*       out = (float*)d_out;

  char*  ws  = (char*)d_ws;
  size_t off = 0;
  auto alloc = [&](size_t bytes) -> void* {
    void* p = ws + off;
    off = (off + bytes + 255) & ~(size_t)255;
    return p;
  };
  __bf16* whT   = (__bf16*)alloc((size_t)N_NODES * F_OUT * sizeof(__bf16));
  float*  s_src = (float*)alloc((size_t)N_NODES * sizeof(float));
  float*  s_dst = (float*)alloc((size_t)N_NODES * sizeof(float));

  // choose j-segment count S for parallelism, limited by workspace size
  int S = 1;
  const int cands[3] = {8, 4, 2};
  for (int c = 0; c < 3; ++c) {
    const size_t need = off + (size_t)cands[c] * N_NODES * sizeof(float2) +
                        (size_t)cands[c] * N_NODES * F_OUT * sizeof(float);
    if (need <= ws_size) { S = cands[c]; break; }
  }
  float2* stats  = (float2*)alloc((size_t)S * N_NODES * sizeof(float2));
  float*  accbuf =
      (S == 1) ? out
               : (float*)alloc((size_t)S * N_NODES * F_OUT * sizeof(float));

  wh_kernel<<<(N_NODES / 16) * (F_OUT / 16), 32, 0, stream>>>(h, W, whT);
  s_kernel<<<N_NODES / 256, 256, 0, stream>>>(whT, a, s_src, s_dst);
  gat_flash_kernel<<<(N_NODES / ROWS) * S, 32, 0, stream>>>(
      adj, whT, s_src, s_dst, accbuf, stats, S);
  combine_kernel<<<(N_NODES * F_OUT) / 256, 256, 0, stream>>>(accbuf, stats,
                                                              out, S);
}
